// PNAConv_34050500722694
// MI455X (gfx1250) — matrix operand order repaired and checked by reading, not compiled, not run on hardware
//
#include <hip/hip_runtime.h>
#include <hip/hip_bf16.h>
#include <math.h>

// ---------------------------------------------------------------------------
// PNAConv for MI455X (gfx1250, wave32, WMMA)
//   phase 1: fused degree + scatter-add of x and x^2 per destination node
//            (float4 gathers, hardware global_atomic_add_f32)
//   phase 2: fused 7-way GEMM (6 heads + residual) via v_wmma_f32_16x16x32_bf16
//            with degree scalers folded into the A-matrix rows.
// ---------------------------------------------------------------------------

typedef __attribute__((ext_vector_type(16))) __bf16 v16bf;
typedef __attribute__((ext_vector_type(8)))  float  v8f;

#define D_DIM 64
#define EPS 1e-7f
#define TILES_PER_BLOCK 8   // 8 waves * 16 nodes = 128 nodes per block

// ------------------- fused degree + scatter kernel -------------------------
// 16 threads per edge, each owning a float4 of the 64-dim feature.
// x, s1, s2 are all L2-resident (25.6 MB each on 192 MB L2); the pass is
// bound by the 204.8M f32 atomic adds, so minimize instructions around them.
__global__ void pna_scatter_kernel(const float* __restrict__ x,
                                   const int* __restrict__ edge_row,
                                   const int* __restrict__ edge_col,
                                   float* __restrict__ deg,
                                   float* __restrict__ s1,
                                   float* __restrict__ s2,
                                   long long total_q) {
    long long gid = (long long)blockIdx.x * blockDim.x + threadIdx.x;
    if (gid >= total_q) return;
    int e = (int)(gid >> 4);
    int q = (int)(gid & 15);            // quad index: dims 4q .. 4q+3
    int src = edge_col[e];
    int dst = edge_row[e];
    const float4 v = *(const float4*)&x[(long long)src * D_DIM + q * 4];
    long long o = (long long)dst * D_DIM + q * 4;
    __hip_atomic_fetch_add(&s1[o + 0], v.x, __ATOMIC_RELAXED, __HIP_MEMORY_SCOPE_AGENT);
    __hip_atomic_fetch_add(&s1[o + 1], v.y, __ATOMIC_RELAXED, __HIP_MEMORY_SCOPE_AGENT);
    __hip_atomic_fetch_add(&s1[o + 2], v.z, __ATOMIC_RELAXED, __HIP_MEMORY_SCOPE_AGENT);
    __hip_atomic_fetch_add(&s1[o + 3], v.w, __ATOMIC_RELAXED, __HIP_MEMORY_SCOPE_AGENT);
    __hip_atomic_fetch_add(&s2[o + 0], v.x * v.x, __ATOMIC_RELAXED, __HIP_MEMORY_SCOPE_AGENT);
    __hip_atomic_fetch_add(&s2[o + 1], v.y * v.y, __ATOMIC_RELAXED, __HIP_MEMORY_SCOPE_AGENT);
    __hip_atomic_fetch_add(&s2[o + 2], v.z * v.z, __ATOMIC_RELAXED, __HIP_MEMORY_SCOPE_AGENT);
    __hip_atomic_fetch_add(&s2[o + 3], v.w * v.w, __ATOMIC_RELAXED, __HIP_MEMORY_SCOPE_AGENT);
    if (q == 0) {
        __hip_atomic_fetch_add(&deg[dst], 1.0f,
                               __ATOMIC_RELAXED, __HIP_MEMORY_SCOPE_AGENT);
    }
}

// ---------------------------- fused GEMM kernel ----------------------------
// Per 16-node tile (one wave): build bf16 A fragments for
//   g=0..2 : mean * {1, logd/avg, avg/(logd+eps)}
//   g=3..5 : std  * {1, logd/avg, avg/(logd+eps)}
//   g=6    : x    (residual, factor 1)
// and accumulate all of them into one f32 C per 16-column tile:
//   acc[t] = sum_g  A_g(16x64) @ W_g^T(64x16-col-tile)   (two K=32 chunks)
__global__ __launch_bounds__(256)
void pna_gemm_kernel(const float* __restrict__ x,
                     const float* __restrict__ pre_w,   // [6][64][64] (o,d)
                     const float* __restrict__ lin_w,   // [64][64]    (o,d)
                     const float* __restrict__ bias,    // [64]
                     const float* __restrict__ avg_ptr, // scalar
                     const float* __restrict__ deg,     // [N]
                     const float* __restrict__ s1,      // [N][64]
                     const float* __restrict__ s2,      // [N][64]
                     float* __restrict__ out,           // [N][64]
                     int n_nodes) {
    // Stage all 7 weight matrices in LDS as bf16: 7*64*64*2 = 57344 B.
    __shared__ __attribute__((aligned(32))) __bf16 wlds[7 * D_DIM * D_DIM];
    for (int idx = threadIdx.x; idx < 7 * D_DIM * D_DIM; idx += 256) {
        float w = (idx < 6 * D_DIM * D_DIM) ? pre_w[idx]
                                            : lin_w[idx - 6 * D_DIM * D_DIM];
        wlds[idx] = (__bf16)w;
    }
    __syncthreads();

    int wave = threadIdx.x >> 5;
    int lane = threadIdx.x & 31;
    int tile = blockIdx.x * TILES_PER_BLOCK + wave;
    int tile_base = tile * 16;
    if (tile_base >= n_nodes) return;   // wave-uniform: EXEC stays all-ones

    int row  = lane & 15;               // A-matrix row this lane owns
    int half = lane >> 4;               // lane group (K sub-block select)
    int node_a = tile_base + row;
    if (node_a >= n_nodes) node_a = n_nodes - 1;   // clamp (addresses only)

    // Degree scalers: bf16 inputs downstream -> fast v_rcp is plenty.
    float avgl = avg_ptr[0];
    float dg   = deg[node_a];
    float logd = __logf(dg + 1.0f);
    float fs0 = 1.0f;
    float fs1 = logd * __builtin_amdgcn_rcpf(avgl);
    float fs2 = avgl * __builtin_amdgcn_rcpf(logd + EPS);

    v8f acc[4];
#pragma unroll
    for (int t = 0; t < 4; ++t) acc[t] = (v8f){0, 0, 0, 0, 0, 0, 0, 0};

#pragma unroll
    for (int kc = 0; kc < 64; kc += 32) {
        // Build the 7 A fragments for this K-chunk.
        // 16-bit A 16x32 layout used here (contiguous k): lane(row, half); element i holds
        //   k = half*16 + i
        v16bf afr[7];
#pragma unroll
        for (int i = 0; i < 16; ++i) {
            int k = kc + (half << 4) + i;
            long long o = (long long)node_a * D_DIM + k;
            float a  = s1[o];
            float b  = s2[o];
            float xv = x[o];
            float var = b - a * a;
            var = var > 0.0f ? var : 0.0f;
            float sd = sqrtf(var + EPS);
            afr[0][i] = (__bf16)(a * fs0);
            afr[1][i] = (__bf16)(a * fs1);
            afr[2][i] = (__bf16)(a * fs2);
            afr[3][i] = (__bf16)(sd * fs0);
            afr[4][i] = (__bf16)(sd * fs1);
            afr[5][i] = (__bf16)(sd * fs2);
            afr[6][i] = (__bf16)xv;
        }
#pragma unroll
        for (int t = 0; t < 4; ++t) {
            int col = (lane & 15) + t * 16;   // B-matrix column this lane owns
            // 16-bit B 32x16 layout used here (same as A): element i holds k = half*16 + i
            // -> 16 consecutive bf16 in [o][d]-major LDS = one 32B ds load.
#pragma unroll
            for (int g = 0; g < 7; ++g) {
                const v16bf bfr = *(const v16bf*)&wlds[(g * D_DIM + col) * D_DIM
                                                       + kc + half * 16];
                acc[t] = __builtin_amdgcn_wmma_f32_16x16x32_bf16(
                    false, afr[g], false, bfr, (short)0, acc[t], false, false);
            }
        }
    }

    // Store: C/D layout: lane(col = lane&15 + 16t), VGPR r -> row half*8 + r.
#pragma unroll
    for (int t = 0; t < 4; ++t) {
        int col = (lane & 15) + t * 16;
        float bv = bias[col];
#pragma unroll
        for (int r = 0; r < 8; ++r) {
            int node_o = tile_base + half * 8 + r;
            if (node_o < n_nodes)
                out[(long long)node_o * D_DIM + col] = acc[t][r] + bv;
        }
    }
}

// ---------------------------------------------------------------------------
extern "C" void kernel_launch(void* const* d_in, const int* in_sizes, int n_in,
                              void* d_out, int out_size, void* d_ws, size_t ws_size,
                              hipStream_t stream) {
    const float* x        = (const float*)d_in[0];
    const int*   edge_row = (const int*)d_in[1];
    const int*   edge_col = (const int*)d_in[2];
    const float* pre_w    = (const float*)d_in[3];
    const float* lin_w    = (const float*)d_in[4];
    const float* bias     = (const float*)d_in[5];
    const float* avg_log  = (const float*)d_in[6];
    float*       out      = (float*)d_out;

    int n_nodes = in_sizes[0] / D_DIM;
    int n_edges = in_sizes[1];

    // Scratch layout: deg [N] | s1 [N*64] | s2 [N*64]  (all f32, zeroed)
    int n_pad = (n_nodes + 255) & ~255;
    float* deg = (float*)d_ws;
    float* s1  = deg + n_pad;
    float* s2  = s1 + (size_t)n_nodes * D_DIM;
    size_t zero_bytes = ((size_t)n_pad + 2ull * n_nodes * D_DIM) * sizeof(float);
    hipMemsetAsync(d_ws, 0, zero_bytes, stream);

    // Phase 1: fused degree + scatter-add (16 threads / edge, float4 gathers)
    {
        long long total_q = (long long)n_edges * 16;
        int threads = 256;
        long long blocks = (total_q + threads - 1) / threads;
        pna_scatter_kernel<<<(int)blocks, threads, 0, stream>>>(
            x, edge_row, edge_col, deg, s1, s2, total_q);
    }
    // Phase 2: fused 7-way WMMA GEMM + scalers + bias
    {
        int tiles  = (n_nodes + 15) / 16;
        int blocks = (tiles + TILES_PER_BLOCK - 1) / TILES_PER_BLOCK;
        pna_gemm_kernel<<<blocks, 256, 0, stream>>>(
            x, pre_w, lin_w, bias, avg_log, deg, s1, s2, out, n_nodes);
    }
}
